// GAT_35820027248975
// MI455X (gfx1250) — matrix operand order repaired and checked by reading, not compiled, not run on hardware
//
#include <hip/hip_runtime.h>
#include <hip/hip_bf16.h>
#include <math.h>

// ---------------------------------------------------------------------------
// 2-layer GAT on gfx1250 (MI455X). Sparse row-softmax formulation of the
// reference's dense NxN attention (NEG_INF entries underflow to exactly 0).
// GEMMs use v_wmma_f32_16x16x32_f16 (K=32 == F_IN in one instruction).
// Aggregation is gather-based over a CSR built on device (no float atomics).
// ---------------------------------------------------------------------------

#define GN      4096      // nodes
#define GE      131072    // input edges
#define GET     (GE + GN) // edges + self loops
#define FIN     32
#define FH      32
#define NH      8
#define NC      10
#define HID     (NH * FH) // 256
#define GALPHA  0.2f

typedef __attribute__((ext_vector_type(16))) _Float16 v16h;
typedef __attribute__((ext_vector_type(8)))  float    v8f;

// ---- fragment K maps used here (contiguous k order, wave32) ----------------------------------
// A 16x32 f16: lane<16 holds row M=lane, K=0..15; lane>=16 holds
// row M=lane-16, K=16..31.
__device__ __forceinline__ int kmapA(int lane, int j) {
    return ((lane & 16) ? 16 : 0) + j;
}
// B 32x16 f16: lane&15 = column N; lane<16 holds K=0..15, lane>=16 K=16..31.
__device__ __forceinline__ int kmapB(int lane, int j) {
    return ((lane & 16) ? 16 : 0) + j;
}

__device__ __forceinline__ float wave_max(float v) {
    #pragma unroll
    for (int o = 16; o > 0; o >>= 1) v = fmaxf(v, __shfl_xor(v, o, 32));
    return v;
}
__device__ __forceinline__ float wave_sum(float v) {
    #pragma unroll
    for (int o = 16; o > 0; o >>= 1) v += __shfl_xor(v, o, 32);
    return v;
}
__device__ __forceinline__ float lrelu(float v) { return v > 0.f ? v : GALPHA * v; }

__device__ __forceinline__ void edge_at(const int* ei, int i, int& s, int& t) {
    if (i < GE) { s = ei[i]; t = ei[GE + i]; }
    else        { s = i - GE; t = i - GE; }   // appended self loops
}

// ---- CSR construction ------------------------------------------------------
__global__ void k_init(int* deg, int* fill) {
    int i = blockIdx.x * blockDim.x + threadIdx.x;
    if (i < GN) { deg[i] = 0; fill[i] = 0; }
}

__global__ void k_count(const int* ei, int* deg) {
    int i = blockIdx.x * blockDim.x + threadIdx.x;
    if (i >= GET) return;
    int s, t; edge_at(ei, i, s, t);
    atomicAdd(&deg[s], 1);
}

__global__ void k_scan(const int* deg, int* rowptr) { // 1 block, 1024 threads
    __shared__ int part[1024];
    int tI = threadIdx.x;
    int v[4]; int s = 0;
    #pragma unroll
    for (int k = 0; k < 4; ++k) { v[k] = deg[tI * 4 + k]; s += v[k]; }
    part[tI] = s;
    __syncthreads();
    for (int off = 1; off < 1024; off <<= 1) {
        int x = (tI >= off) ? part[tI - off] : 0;
        __syncthreads();
        part[tI] += x;
        __syncthreads();
    }
    int run = (tI == 0) ? 0 : part[tI - 1];
    #pragma unroll
    for (int k = 0; k < 4; ++k) { rowptr[tI * 4 + k] = run; run += v[k]; }
    if (tI == 1023) rowptr[GN] = run;
}

__global__ void k_scatter(const int* ei, const int* rowptr, int* fill, int* col) {
    int i = blockIdx.x * blockDim.x + threadIdx.x;
    if (i >= GET) return;
    int s, t; edge_at(ei, i, s, t);
    int pos = rowptr[s] + atomicAdd(&fill[s], 1);
    col[pos] = t;
}

// ---- Layer 1 GEMM: h[i, hd*32+f] = (x @ W1[hd])[i,f]  (f32 via f16 WMMA) ---
// One wave per (head, 16-row strip); K=32 done in a single WMMA per col-tile.
__global__ void k_gemm1(const float* __restrict__ x, const float* __restrict__ W1,
                        float* __restrict__ h1) {
    int wid  = blockIdx.x * (blockDim.x / 32) + (threadIdx.x >> 5); // 0..2047
    int lane = threadIdx.x & 31;
    int hd   = wid >> 8;            // 0..7
    int m0   = (wid & 255) * 16;    // row base
    const float* W = W1 + hd * (FIN * FH);

    // A fragment: rows m0..m0+15 of x (16x32, f32 -> f16)
    v16h a{};
    int m = m0 + (lane & 15);
    #pragma unroll
    for (int j = 0; j < 16; ++j)
        a[j] = (_Float16)x[m * FIN + kmapA(lane, j)];

    int n = lane & 15;
    #pragma unroll
    for (int ct = 0; ct < 2; ++ct) {            // column tiles 0..15, 16..31
        int n0 = ct * 16;
        v16h b{};
        #pragma unroll
        for (int j = 0; j < 16; ++j)
            b[j] = (_Float16)W[kmapB(lane, j) * FH + (n0 + n)];
        v8f c = {};
        c = __builtin_amdgcn_wmma_f32_16x16x32_f16(false, a, false, b,
                                                   (short)0, c, false, false);
        int rbase = m0 + ((lane & 16) ? 8 : 0);
        #pragma unroll
        for (int r = 0; r < 8; ++r)
            h1[(rbase + r) * HID + hd * FH + n0 + n] = c[r];
    }
}

// ---- el/er projections: el = h . a1[:FH], er = h . a1[FH:] -----------------
__global__ void k_elr1(const float* __restrict__ h1, const float* __restrict__ a1,
                       float* __restrict__ el, float* __restrict__ er) {
    int i = blockIdx.x * blockDim.x + threadIdx.x; // node*NH + hd flat
    if (i >= GN * NH) return;
    int node = i >> 3, hd = i & 7;
    const float* hp = h1 + node * HID + hd * FH;
    const float* ap = a1 + hd * (2 * FH);
    float sl = 0.f, sr = 0.f;
    #pragma unroll
    for (int f = 0; f < FH; ++f) { sl += hp[f] * ap[f]; sr += hp[f] * ap[FH + f]; }
    el[hd * GN + node] = sl;
    er[hd * GN + node] = sr;
}

// ---- Layer 1 row softmax + aggregate + bias + ELU -> x1 (f16) --------------
// One wave per (node, head); lane = feature.
__global__ void k_row1(const int* __restrict__ rowptr, const int* __restrict__ col,
                       const float* __restrict__ h1, const float* __restrict__ el,
                       const float* __restrict__ er, const float* __restrict__ b1,
                       _Float16* __restrict__ x1h) {
    int wid  = blockIdx.x * (blockDim.x / 32) + (threadIdx.x >> 5);
    int lane = threadIdx.x & 31;
    int node = wid >> 3, hd = wid & 7;
    if (node >= GN) return;
    int start = rowptr[node], end = rowptr[node + 1];
    float eli = el[hd * GN + node];

    float mx = -1e30f;
    for (int j = start + lane; j < end; j += 32)
        mx = fmaxf(mx, lrelu(eli + er[hd * GN + col[j]]));
    mx = wave_max(mx);
    float sm = 0.f;
    for (int j = start + lane; j < end; j += 32)
        sm += expf(lrelu(eli + er[hd * GN + col[j]]) - mx);
    sm = wave_sum(sm);
    float inv = 1.0f / sm;

    float acc = 0.f;
    for (int j = start; j < end; ++j) {
        int t = col[j];                                   // uniform across wave
        float p = expf(lrelu(eli + er[hd * GN + t]) - mx) * inv;
        acc += p * h1[t * HID + hd * FH + lane];          // coalesced gather
    }
    float v = acc + b1[hd * FH + lane];
    v = v > 0.f ? v : expm1f(v);                          // ELU
    x1h[node * HID + hd * FH + lane] = (_Float16)v;
}

// ---- Layer 2 GEMM: h2 = x1 @ W2  (K=256 -> 8 chained WMMAs, N padded to 16)
__global__ void k_gemm2(const _Float16* __restrict__ x1h, const float* __restrict__ W2,
                        float* __restrict__ h2) {
    int wid  = blockIdx.x * (blockDim.x / 32) + (threadIdx.x >> 5); // 0..255
    int lane = threadIdx.x & 31;
    int m0   = wid * 16;
    int m    = m0 + (lane & 15);
    int n    = lane & 15;

    v8f c = {};
    #pragma unroll
    for (int kk = 0; kk < 8; ++kk) {
        v16h a{}, b{};
        #pragma unroll
        for (int j = 0; j < 16; ++j)
            a[j] = x1h[m * HID + kk * 32 + kmapA(lane, j)];
        #pragma unroll
        for (int j = 0; j < 16; ++j) {
            int k = kk * 32 + kmapB(lane, j);
            b[j] = (n < NC) ? (_Float16)W2[k * NC + n] : (_Float16)0.f;
        }
        c = __builtin_amdgcn_wmma_f32_16x16x32_f16(false, a, false, b,
                                                   (short)0, c, false, false);
    }
    int rbase = m0 + ((lane & 16) ? 8 : 0);
    #pragma unroll
    for (int r = 0; r < 8; ++r)
        h2[(rbase + r) * 16 + n] = c[r];   // cols 10..15 are exact zeros
}

__global__ void k_elr2(const float* __restrict__ h2, const float* __restrict__ a2,
                       float* __restrict__ el2, float* __restrict__ er2) {
    int i = blockIdx.x * blockDim.x + threadIdx.x;
    if (i >= GN) return;
    float sl = 0.f, sr = 0.f;
    #pragma unroll
    for (int c = 0; c < NC; ++c) {
        float h = h2[i * 16 + c];
        sl += h * a2[c];
        sr += h * a2[NC + c];
    }
    el2[i] = sl; er2[i] = sr;
}

// ---- Layer 2 row softmax + aggregate + bias + log_softmax ------------------
__global__ void k_row2(const int* __restrict__ rowptr, const int* __restrict__ col,
                       const float* __restrict__ h2, const float* __restrict__ el2,
                       const float* __restrict__ er2, const float* __restrict__ b2,
                       float* __restrict__ out) {
    int wid  = blockIdx.x * (blockDim.x / 32) + (threadIdx.x >> 5);
    int lane = threadIdx.x & 31;
    int node = wid;
    if (node >= GN) return;
    int start = rowptr[node], end = rowptr[node + 1];
    float eli = el2[node];

    float mx = -1e30f;
    for (int j = start + lane; j < end; j += 32)
        mx = fmaxf(mx, lrelu(eli + er2[col[j]]));
    mx = wave_max(mx);
    float sm = 0.f;
    for (int j = start + lane; j < end; j += 32)
        sm += expf(lrelu(eli + er2[col[j]]) - mx);
    sm = wave_sum(sm);
    float inv = 1.0f / sm;

    float acc = 0.f;
    bool live = lane < 16;
    for (int j = start; j < end; ++j) {
        int t = col[j];
        float p = expf(lrelu(eli + er2[t]) - mx) * inv;
        if (live) acc += p * h2[t * 16 + lane];
    }
    float o = acc + ((lane < NC) ? b2[lane] : 0.f);
    // log_softmax over 10 classes across lanes 0..9
    float mv = (lane < NC) ? o : -1e30f;
    mv = wave_max(mv);
    float ex = (lane < NC) ? expf(o - mv) : 0.f;
    float se = wave_sum(ex);
    if (lane < NC) out[node * NC + lane] = o - mv - logf(se);
}

// ---------------------------------------------------------------------------
extern "C" void kernel_launch(void* const* d_in, const int* in_sizes, int n_in,
                              void* d_out, int out_size, void* d_ws, size_t ws_size,
                              hipStream_t stream) {
    (void)in_sizes; (void)n_in; (void)out_size; (void)ws_size;
    const float* x   = (const float*)d_in[0];
    const int*   ei  = (const int*)  d_in[1];
    const float* W1  = (const float*)d_in[2];
    const float* a1  = (const float*)d_in[3];
    const float* b1  = (const float*)d_in[4];
    const float* W2  = (const float*)d_in[5];
    const float* a2  = (const float*)d_in[6];
    const float* b2  = (const float*)d_in[7];
    float* out = (float*)d_out;

    // workspace carve-out (256B aligned slices)
    char* ws = (char*)d_ws;
    size_t o = 0;
    auto take = [&](size_t bytes) -> char* {
        char* p = ws + o;
        o = (o + bytes + 255) & ~(size_t)255;
        return p;
    };
    int*      rowptr = (int*)     take((GN + 1) * sizeof(int));
    int*      deg    = (int*)     take(GN * sizeof(int));
    int*      fill   = (int*)     take(GN * sizeof(int));
    int*      col    = (int*)     take(GET * sizeof(int));
    float*    h1     = (float*)   take((size_t)GN * HID * sizeof(float));
    float*    el1    = (float*)   take((size_t)NH * GN * sizeof(float));
    float*    er1    = (float*)   take((size_t)NH * GN * sizeof(float));
    _Float16* x1h    = (_Float16*)take((size_t)GN * HID * sizeof(_Float16));
    float*    h2     = (float*)   take((size_t)GN * 16 * sizeof(float));
    float*    el2    = (float*)   take(GN * sizeof(float));
    float*    er2    = (float*)   take(GN * sizeof(float));

    const int TB = 256;
    // CSR build
    k_init   <<<(GN + TB - 1) / TB, TB, 0, stream>>>(deg, fill);
    k_count  <<<(GET + TB - 1) / TB, TB, 0, stream>>>(ei, deg);
    k_scan   <<<1, 1024, 0, stream>>>(deg, rowptr);
    k_scatter<<<(GET + TB - 1) / TB, TB, 0, stream>>>(ei, rowptr, fill, col);
    // layer 1
    k_gemm1  <<<(NH * (GN / 16)) / (TB / 32), TB, 0, stream>>>(x, W1, h1);
    k_elr1   <<<(GN * NH + TB - 1) / TB, TB, 0, stream>>>(h1, a1, el1, er1);
    k_row1   <<<(GN * NH) / (TB / 32), TB, 0, stream>>>(rowptr, col, h1, el1, er1, b1, x1h);
    // layer 2
    k_gemm2  <<<(GN / 16) / (TB / 32), TB, 0, stream>>>(x1h, W2, h2);
    k_elr2   <<<(GN + TB - 1) / TB, TB, 0, stream>>>(h2, a2, el2, er2);
    k_row2   <<<GN / (TB / 32), TB, 0, stream>>>(rowptr, col, h2, el2, er2, b2, out);
}
